// TransformerBlock_90443421319256
// MI455X (gfx1250) — matrix operand order repaired and checked by reading, not compiled, not run on hardware
//
#include <hip/hip_runtime.h>
#include <stdint.h>

// ---------------------------------------------------------------------------
// CDNA5 (gfx1250) transformer block: RMSNorm -> GEMM1 -> RoPE -> local attn
// -> GELU -> GEMM2.  All matmuls via v_wmma_f32_16x16x32_bf16 (wave32).
// Global->LDS A-tile staging uses gfx1250 async-to-LDS DMA (ASYNCcnt).
// ---------------------------------------------------------------------------

typedef __bf16        v16bf __attribute__((ext_vector_type(16)));
typedef float         v8f   __attribute__((ext_vector_type(8)));
typedef unsigned int  v4u   __attribute__((ext_vector_type(4)));

#define HIDDEN   768
#define INTER    3072
#define NHEADS   12
#define HEAD     64
#define NTOK     16384
#define SEQLEN   2048
#define WATTN    512
#define COLS_IN  (3*HIDDEN + INTER)   // 5376
#define COLS_CMB (HIDDEN + INTER)     // 3840

#define USE_ASYNC_G2L 1

__device__ __forceinline__ unsigned short f2bf(float f) {
  unsigned u = __float_as_uint(f);
  u += 0x7fffu + ((u >> 16) & 1u);          // round-to-nearest-even
  return (unsigned short)(u >> 16);
}

union Frag { v16bf v; v4u q[2]; };

__device__ __forceinline__ v8f zero8() {
  v8f z;
#pragma unroll
  for (int i = 0; i < 8; ++i) z[i] = 0.0f;
  return z;
}

__device__ __forceinline__ v8f wmma_bf16(const Frag& a, const Frag& b, v8f c) {
  return __builtin_amdgcn_wmma_f32_16x16x32_bf16(false, a.v, false, b.v,
                                                 (short)0, c, false, false);
}

// Async global->LDS 16B copy.  VDST carries the per-lane LDS byte offset
// (low 32 bits of the generic LDS address per the aperture mapping); the
// hardware adds LDS_BASE.  Tracked by ASYNCcnt.
__device__ __forceinline__ void g2l_b128(const unsigned short* g,
                                         unsigned short* l) {
#if USE_ASYNC_G2L
  unsigned loff = (unsigned)(size_t)l;
  asm volatile("global_load_async_to_lds_b128 %0, %1, off"
               :: "v"(loff), "v"(g) : "memory");
#else
  *(v4u*)l = *(const v4u*)g;
#endif
}
__device__ __forceinline__ void g2l_commit() {
#if USE_ASYNC_G2L
  asm volatile("s_wait_asynccnt 0x0" ::: "memory");
#endif
}

// ---------------------------------------------------------------------------
// fp32 -> bf16 elementwise (weights)
// ---------------------------------------------------------------------------
__global__ void k_f32_to_bf16(const float* __restrict__ src,
                              unsigned short* __restrict__ dst, int n) {
  int i = blockIdx.x * 256 + threadIdx.x;
  if (i < n) dst[i] = f2bf(src[i]);
}

// ---------------------------------------------------------------------------
// RMSNorm (one row per workgroup) + cast to bf16
// ---------------------------------------------------------------------------
__global__ void k_rmsnorm(const float* __restrict__ x,
                          const float* __restrict__ scale,
                          unsigned short* __restrict__ xb) {
  __shared__ float red[256];
  int row = blockIdx.x, tid = threadIdx.x;
  const float* xr = x + (size_t)row * HIDDEN;
  float s = 0.f;
  for (int c = tid; c < HIDDEN; c += 256) { float v = xr[c]; s += v * v; }
  red[tid] = s;
  __syncthreads();
  for (int off = 128; off > 0; off >>= 1) {
    if (tid < off) red[tid] += red[tid + off];
    __syncthreads();
  }
  float rn = rsqrtf(red[0] * (1.0f / HIDDEN) + 1e-5f);
  for (int c = tid; c < HIDDEN; c += 256)
    xb[(size_t)row * HIDDEN + c] = f2bf(xr[c] * rn * scale[c]);
}

// ---------------------------------------------------------------------------
// bf16 GEMM: C[M,N] = A[M,K] * B[K,N] + bias[N]   (fp32 accumulate/output)
// 128x128x32 workgroup tile, 8 waves, wave = 64x32 (4x2 wmma tiles).
// A tile staged with async-to-LDS DMA; B tile transposed through registers
// (two b128 loads batched, then b16 scatter) for k-contiguous fragments.
// ---------------------------------------------------------------------------
__global__ void __launch_bounds__(256)
k_gemm_bf16(const unsigned short* __restrict__ A,
            const unsigned short* __restrict__ B,
            const float* __restrict__ bias,
            float* __restrict__ C, int M, int N, int K) {
  __shared__ unsigned short sA[128][32];   // [m][k]
  __shared__ unsigned short sB[128][32];   // transposed: [n][k]
  (void)M;
  int tid = threadIdx.x;
  int lane = tid & 31, wid = tid >> 5;
  int ln = lane & 15, lh = lane >> 4;
  int waveM = wid & 1, waveN = wid >> 1;
  int m0 = blockIdx.y * 128, n0 = blockIdx.x * 128;

  // staging maps: A -> thread t covers sA[t>>1][(t&1)*16 .. +15] (32B)
  //               B -> thread t covers k-row t>>3, 16 n's at (t&7)*16
  int ar = tid >> 1, ac = (tid & 1) << 4;
  int bk = tid >> 3, bn = (tid & 7) << 4;

  v8f acc[4][2];
#pragma unroll
  for (int i = 0; i < 4; ++i)
#pragma unroll
    for (int j = 0; j < 2; ++j) acc[i][j] = zero8();

  int nk = K >> 5;
  for (int kt = 0; kt < nk; ++kt) {
    int k0 = kt << 5;

    // --- stage A tile: async DMA, no VGPR round-trip ---
    const unsigned short* gA = A + (size_t)(m0 + ar) * K + k0 + ac;
    g2l_b128(gA,     &sA[ar][ac]);
    g2l_b128(gA + 8, &sA[ar][ac + 8]);

    // --- stage B tile transposed: batch both wide loads, then scatter ---
    const unsigned short* gB = B + (size_t)(k0 + bk) * N + n0 + bn;
    union { v4u q; unsigned short h[8]; } u0, u1;
    u0.q = *(const v4u*)gB;
    u1.q = *(const v4u*)(gB + 8);
#pragma unroll
    for (int e = 0; e < 8; ++e) {
      sB[bn + e][bk]     = u0.h[e];
      sB[bn + 8 + e][bk] = u1.h[e];
    }

    if (kt + 1 < nk) {   // hint next K tile toward L2 (global_prefetch_b8)
      __builtin_prefetch(gA + 32, 0, 0);
      __builtin_prefetch(gB + (size_t)32 * N, 0, 0);
    }

    g2l_commit();        // s_wait_asynccnt 0
    __syncthreads();

    Frag af[4], bf[2];
#pragma unroll
    for (int ms = 0; ms < 4; ++ms) {
      int row = waveM * 64 + ms * 16 + ln;
      int c0 = lh * 16;                // lanes 0-15: K 0-15, hi: K 16-31
      af[ms].q[0] = *(const v4u*)&sA[row][c0];
      af[ms].q[1] = *(const v4u*)&sA[row][c0 + 8];
    }
#pragma unroll
    for (int ns = 0; ns < 2; ++ns) {
      int col = waveN * 32 + ns * 16 + ln;
      int kk = lh * 16;                // lanes 0-15: K 0-15, hi: K 16-31
      bf[ns].q[0] = *(const v4u*)&sB[col][kk];
      bf[ns].q[1] = *(const v4u*)&sB[col][kk + 8];
    }
#pragma unroll
    for (int ms = 0; ms < 4; ++ms)
#pragma unroll
      for (int ns = 0; ns < 2; ++ns)
        acc[ms][ns] = wmma_bf16(af[ms], bf[ns], acc[ms][ns]);
    __syncthreads();
  }

  // epilogue: C-tile layout -> VGPR j holds M=j (+8 for hi lanes), N=lane&15
#pragma unroll
  for (int ns = 0; ns < 2; ++ns) {
    int n = n0 + waveN * 32 + ns * 16 + ln;
    float bv = bias[n];
#pragma unroll
    for (int ms = 0; ms < 4; ++ms) {
      int mbase = m0 + waveM * 64 + ms * 16 + lh * 8;
#pragma unroll
      for (int j = 0; j < 8; ++j)
        C[(size_t)(mbase + j) * N + n] = acc[ms][ns][j] + bv;
    }
  }
}

// ---------------------------------------------------------------------------
// RoPE on q,k (bf16 out) + v cast + GELU(ff) into combined[:,768:]
// ---------------------------------------------------------------------------
__global__ void k_rope_gelu(const float* __restrict__ mid,
                            const float* __restrict__ psin,
                            const float* __restrict__ pcos,
                            unsigned short* __restrict__ qb,
                            unsigned short* __restrict__ kb,
                            unsigned short* __restrict__ vb,
                            unsigned short* __restrict__ comb) {
  int i = blockIdx.x, tid = threadIdx.x;
  const float* row = mid + (size_t)i * COLS_IN;
  const float* sr = psin + (size_t)i * HIDDEN;
  const float* cr = pcos + (size_t)i * HIDDEN;
  for (int p = tid; p < HIDDEN / 2; p += 256) {
    int d = p * 2;
    float s0 = sr[d], s1 = sr[d + 1], c0 = cr[d], c1 = cr[d + 1];
    float qe = row[d], qo = row[d + 1];
    qb[(size_t)i * HIDDEN + d]     = f2bf(qe * c0 - qo * s0);
    qb[(size_t)i * HIDDEN + d + 1] = f2bf(qo * c1 + qe * s1);
    float ke = row[HIDDEN + d], ko = row[HIDDEN + d + 1];
    kb[(size_t)i * HIDDEN + d]     = f2bf(ke * c0 - ko * s0);
    kb[(size_t)i * HIDDEN + d + 1] = f2bf(ko * c1 + ke * s1);
  }
  for (int d = tid; d < HIDDEN; d += 256)
    vb[(size_t)i * HIDDEN + d] = f2bf(row[2 * HIDDEN + d]);
  for (int d = tid; d < INTER; d += 256) {
    float v = row[3 * HIDDEN + d];
    float g = 0.5f * v * (1.0f + tanhf(0.7978845608028654f * (v + 0.044715f * v * v * v)));
    comb[(size_t)i * COLS_CMB + HIDDEN + d] = f2bf(g);
  }
}

// ---------------------------------------------------------------------------
// Local attention: one WG per (head, 16-query-row tile).  Keys = prev+cur
// 512-block (1024 cols).  WMMA for QK^T and P*V, exact softmax in between.
// Dynamic LDS: S fp32 16x1024 | P bf16 16x1024 | Vt bf16 64x1024 | reductions
// ---------------------------------------------------------------------------
__global__ void __launch_bounds__(256)
k_attn(const unsigned short* __restrict__ qb,
       const unsigned short* __restrict__ kb,
       const unsigned short* __restrict__ vb,
       unsigned short* __restrict__ comb) {
  extern __shared__ char smem[];
  float*          S  = (float*)smem;                                   // 64 KB
  unsigned short* P  = (unsigned short*)(smem + 65536);                // 32 KB
  unsigned short* Vt = (unsigned short*)(smem + 65536 + 32768);        // 128 KB
  float* red   = (float*)(smem + 65536 + 32768 + 131072);              // 256 f
  float* rowst = red + 256;                                            // 32 f

  int tid = threadIdx.x, lane = tid & 31, wid = tid >> 5;
  int ln = lane & 15, lh = lane >> 4;
  int rt = blockIdx.x, h = blockIdx.y;
  int qrow0 = rt << 4;
  int blk = qrow0 >> 9;
  int key0 = (blk << 9) - WATTN;            // global pos of score column 0

  // ---- phase 1: S = mask(Q K^T * 1/sqrt(64)) --------------------------------
  const unsigned short* qbase = qb + (size_t)(qrow0 + ln) * HIDDEN + h * HEAD;
#pragma unroll
  for (int t = 0; t < 8; ++t) {
    int nt = wid * 8 + t;
    int colbase = nt * 16;
    int ktok = key0 + colbase + ln;
    int ktokc = ktok < 0 ? 0 : ktok;        // clamp; mask kills the value
    const unsigned short* kbase = kb + (size_t)ktokc * HIDDEN + h * HEAD;
    v8f acc = zero8();
#pragma unroll
    for (int kh = 0; kh < 64; kh += 32) {
      Frag a, b;
      int c0 = kh + lh * 16;
      a.q[0] = *(const v4u*)(qbase + c0);
      a.q[1] = *(const v4u*)(qbase + c0 + 8);
      int d0 = kh + lh * 16;
      b.q[0] = *(const v4u*)(kbase + d0);
      b.q[1] = *(const v4u*)(kbase + d0 + 8);
      acc = wmma_bf16(a, b, acc);
    }
    int jj = key0 + colbase + ln;
    bool iscur = (colbase + ln) >= WATTN;
#pragma unroll
    for (int j = 0; j < 8; ++j) {
      int m = lh * 8 + j;
      int i = qrow0 + m;
      bool valid = iscur ? (jj <= i)
                         : (jj >= 0 && (i - jj) < WATTN &&
                            (i / SEQLEN) == (jj / SEQLEN));
      S[m * 1024 + colbase + ln] = valid ? acc[j] * 0.125f : -3.0e38f;
    }
  }
  __syncthreads();

  // ---- phase 1.5: stage V^T (64 dims x 1024 keys); batch loads, then scatter
  for (int it = 0; it < 4; ++it) {
    int c = tid + it * 256;
    int tok = key0 + c;
    int tokc = tok < 0 ? 0 : tok;
    const unsigned short* src = vb + (size_t)tokc * HIDDEN + h * HEAD;
    union { v4u q; unsigned short hh[8]; } u[8];
#pragma unroll
    for (int d8 = 0; d8 < 8; ++d8) u[d8].q = *(const v4u*)(src + d8 * 8);
#pragma unroll
    for (int d8 = 0; d8 < 8; ++d8)
#pragma unroll
      for (int e = 0; e < 8; ++e)
        Vt[(d8 * 8 + e) * 1024 + c] = u[d8].hh[e];
  }

  // ---- phase 2: exact softmax over 1024 cols --------------------------------
  {
    int r = tid >> 4, g = tid & 15;
    float mx = -3.4e38f;
    for (int c = g; c < 1024; c += 16) mx = fmaxf(mx, S[r * 1024 + c]);
    red[r * 16 + g] = mx;
    __syncthreads();
    if (g == 0) {
      float m2 = red[r * 16];
      for (int t = 1; t < 16; ++t) m2 = fmaxf(m2, red[r * 16 + t]);
      rowst[r] = m2;
    }
    __syncthreads();
    float rm = rowst[r];
    float sum = 0.f;
    for (int c = g; c < 1024; c += 16) {
      float e = __expf(S[r * 1024 + c] - rm);
      S[r * 1024 + c] = e;
      sum += e;
    }
    red[r * 16 + g] = sum;
    __syncthreads();
    if (g == 0) {
      float s2 = 0.f;
      for (int t = 0; t < 16; ++t) s2 += red[r * 16 + t];
      rowst[16 + r] = s2;
    }
    __syncthreads();
    float inv = 1.0f / rowst[16 + r];
    for (int c = g; c < 1024; c += 16)
      P[r * 1024 + c] = f2bf(S[r * 1024 + c] * inv);
  }
  __syncthreads();

  // ---- phase 3: O = P (16x1024) * V (1024x64); wave w owns dims w*16..+15 ---
  if (wid < 4) {                            // wave-uniform branch: EXEC all-1s
    v8f o = zero8();
    for (int kc = 0; kc < 32; ++kc) {
      Frag a, b;
      int c0 = kc * 32 + lh * 16;
      a.q[0] = *(const v4u*)&P[ln * 1024 + c0];
      a.q[1] = *(const v4u*)&P[ln * 1024 + c0 + 8];
      int dd = wid * 16 + ln;
      int k0 = kc * 32 + lh * 16;
      b.q[0] = *(const v4u*)&Vt[dd * 1024 + k0];
      b.q[1] = *(const v4u*)&Vt[dd * 1024 + k0 + 8];
      o = wmma_bf16(a, b, o);
    }
    int n = wid * 16 + ln;
#pragma unroll
    for (int j = 0; j < 8; ++j) {
      int tok = qrow0 + lh * 8 + j;
      comb[(size_t)tok * COLS_CMB + h * HEAD + n] = f2bf(o[j]);
    }
  }
}

// ---------------------------------------------------------------------------
extern "C" void kernel_launch(void* const* d_in, const int* in_sizes, int n_in,
                              void* d_out, int out_size, void* d_ws, size_t ws_size,
                              hipStream_t stream) {
  (void)in_sizes; (void)n_in; (void)out_size; (void)ws_size;
  const float* x      = (const float*)d_in[0];
  /* d_in[1] normed_ages: unused by reference */
  const float* psin   = (const float*)d_in[2];
  const float* pcos   = (const float*)d_in[3];
  const float* rscale = (const float*)d_in[4];
  const float* W_in   = (const float*)d_in[5];
  const float* b_in   = (const float*)d_in[6];
  const float* W_out  = (const float*)d_in[7];
  const float* b_out  = (const float*)d_in[8];
  float* out = (float*)d_out;

  char* ws = (char*)d_ws;
  size_t off = 0;
  auto alloc = [&](size_t bytes) {
    char* p = ws + off;
    off += (bytes + 255) & ~(size_t)255;
    return p;
  };
  unsigned short* Xb    = (unsigned short*)alloc((size_t)NTOK * HIDDEN * 2);
  unsigned short* Wbin  = (unsigned short*)alloc((size_t)HIDDEN * COLS_IN * 2);
  unsigned short* Wbout = (unsigned short*)alloc((size_t)COLS_CMB * HIDDEN * 2);
  float*          mid   = (float*)alloc((size_t)NTOK * COLS_IN * 4);
  unsigned short* qbuf  = (unsigned short*)alloc((size_t)NTOK * HIDDEN * 2);
  unsigned short* kbuf  = (unsigned short*)alloc((size_t)NTOK * HIDDEN * 2);
  unsigned short* vbuf  = (unsigned short*)alloc((size_t)NTOK * HIDDEN * 2);
  unsigned short* comb  = (unsigned short*)alloc((size_t)NTOK * COLS_CMB * 2);

  int nWin = HIDDEN * COLS_IN;
  k_f32_to_bf16<<<(nWin + 255) / 256, 256, 0, stream>>>(W_in, Wbin, nWin);
  int nWout = COLS_CMB * HIDDEN;
  k_f32_to_bf16<<<(nWout + 255) / 256, 256, 0, stream>>>(W_out, Wbout, nWout);

  k_rmsnorm<<<NTOK, 256, 0, stream>>>(x, rscale, Xb);

  k_gemm_bf16<<<dim3(COLS_IN / 128, NTOK / 128), 256, 0, stream>>>(
      Xb, Wbin, b_in, mid, NTOK, COLS_IN, HIDDEN);

  k_rope_gelu<<<NTOK, 256, 0, stream>>>(mid, psin, pcos, qbuf, kbuf, vbuf, comb);

  size_t attn_smem = 65536 + 32768 + 131072 + 256 * 4 + 32 * 4;  // ~225 KB
  (void)hipFuncSetAttribute((const void*)k_attn,
                            hipFuncAttributeMaxDynamicSharedMemorySize,
                            (int)attn_smem);
  k_attn<<<dim3(NTOK / 16, NHEADS), 256, attn_smem, stream>>>(qbuf, kbuf, vbuf, comb);

  k_gemm_bf16<<<dim3(HIDDEN / 128, NTOK / 128), 256, 0, stream>>>(
      comb, Wbout, b_out, out, NTOK, HIDDEN, COLS_CMB);
}
